// LinearAttention_20529943675310
// MI455X (gfx1250) — matrix operand order repaired and checked by reading, not compiled, not run on hardware
//
#include <hip/hip_runtime.h>
#include <hip/hip_bf16.h>
#include <math.h>

typedef __bf16 bh_t;
typedef __bf16  bf16x16 __attribute__((ext_vector_type(16)));
typedef __bf16  bf16x8  __attribute__((ext_vector_type(8)));
typedef float   f32x8   __attribute__((ext_vector_type(8)));
typedef unsigned int u32x4 __attribute__((ext_vector_type(4)));
typedef int          i32x4 __attribute__((ext_vector_type(4)));
typedef int          i32x8 __attribute__((ext_vector_type(8)));

#define NSEQ   16384
#define NBATCH 8
#define CDIM   128
#define O3     384
#define TPITCH 144   // 128 elems + 8-DWORD pad per row (descriptor pad_amount=7, pad_interval=5)

#if defined(__HIP_DEVICE_COMPILE__) && __has_builtin(__builtin_amdgcn_tensor_load_to_lds)
#define HAVE_TDM 1
#else
#define HAVE_TDM 0
#endif

static __device__ __forceinline__ bf16x16 ld16(const bh_t* p0, const bh_t* p1) {
    bf16x8 a = *(const bf16x8*)p0;
    bf16x8 b = *(const bf16x8*)p1;
    bf16x16 r;
#pragma unroll
    for (int i = 0; i < 8; ++i) { r[i] = a[i]; r[i + 8] = b[i]; }
    return r;
}

static __device__ __forceinline__ f32x8 wmma_bf16(bf16x16 a, bf16x16 b, f32x8 c) {
    return __builtin_amdgcn_wmma_f32_16x16x32_bf16(false, a, false, b, (short)0, c, false, false);
}

static __device__ __forceinline__ unsigned lds_offset(const void* p) {
    // generic -> LDS(addrspace 3) cast; ptrtoint of an AS(3) pointer is the 32-bit LDS offset
    return (unsigned)(unsigned long long)(__attribute__((address_space(3))) const char*)p;
}

// Issue a 2D TDM load: tile_rows rows of 128 bf16 (256B), LDS-padded to 288B pitch.
static __device__ __forceinline__ void tdm_load_tile(unsigned lds_byte_off, const bh_t* gsrc,
                                                     unsigned rows) {
#if HAVE_TDM
    unsigned long long ga = (unsigned long long)gsrc;
    u32x4 g0;
    g0[0] = 1u;                                         // count=1 (valid user descriptor)
    g0[1] = lds_byte_off;                               // lds_addr
    g0[2] = (unsigned)(ga & 0xFFFFFFFFu);               // global_addr[31:0]
    g0[3] = (unsigned)((ga >> 32) & 0x1FFFFFFu) | (2u << 30); // global_addr[56:32] | type=2
    i32x8 g1;
    g1[0] = (int)((1u << 16)            // data_size = 1 -> 2 bytes
                | (1u << 20)            // pad_enable
                | (5u << 22)            // pad_interval code 5 -> every 64 DWORDs (256B row)
                | (7u << 25));          // pad_amount code 7  -> 8 DWORDs (32B) pad
    g1[1] = (int)((CDIM & 0xFFFFu) << 16);              // tensor_dim0[15:0] @ bits 63:48
    g1[2] = (int)(((CDIM >> 16) & 0xFFFFu) | ((rows & 0xFFFFu) << 16)); // dim0 hi | dim1 lo
    g1[3] = (int)(((rows >> 16) & 0xFFFFu) | ((CDIM & 0xFFFFu) << 16)); // dim1 hi | tile_dim0
    g1[4] = (int)(rows & 0xFFFFu);                      // tile_dim1 (tile_dim2 = 0)
    g1[5] = (int)CDIM;                                  // tensor_dim0_stride[31:0]
    g1[6] = 0;                                          // stride hi, dim1_stride lo
    g1[7] = 0;
    i32x4 z4; z4[0] = 0; z4[1] = 0; z4[2] = 0; z4[3] = 0;
    i32x8 z8;
#pragma unroll
    for (int i = 0; i < 8; ++i) z8[i] = 0;
    __builtin_amdgcn_tensor_load_to_lds(g0, g1, z4, z4, z8, 0);
#else
    (void)lds_byte_off; (void)gsrc; (void)rows;
#endif
}

// stage a [rows x 128] bf16 tile (row-major, contiguous) into padded LDS [rows][TPITCH]
static __device__ __forceinline__ void stage_tile(bh_t dst[][TPITCH], const bh_t* gsrc,
                                                  int rows, int tid) {
#if HAVE_TDM
    if (tid < 32) {   // wave 0 issues the TDM DMA; EXEC is ignored by tensor ops
        tdm_load_tile(lds_offset(&dst[0][0]), gsrc, (unsigned)rows);
        __builtin_amdgcn_s_wait_tensorcnt(0);
    }
#else
    for (int idx = tid; idx < rows * (CDIM / 8); idx += 256) {
        int r = idx / (CDIM / 8), c = (idx % (CDIM / 8)) * 8;
        *(bf16x8*)&dst[r][c] = *(const bf16x8*)(gsrc + (size_t)r * CDIM + c);
    }
#endif
}

// ---------------- conversion kernels ----------------
__global__ __launch_bounds__(256) void k_cvt_xT(const float* __restrict__ x, bh_t* __restrict__ xT) {
    __shared__ float tile[32][33];
    const int b = blockIdx.z, c0 = blockIdx.y * 32, n0 = blockIdx.x * 32;
    const int tid = threadIdx.x;
    for (int i = tid; i < 32 * 32; i += 256) {
        int ci = i >> 5, nj = i & 31;
        tile[ci][nj] = x[((size_t)b * CDIM + c0 + ci) * NSEQ + n0 + nj];
    }
    __syncthreads();
    for (int i = tid; i < 32 * 32; i += 256) {
        int nj = i >> 5, ci = i & 31;
        xT[((size_t)b * NSEQ + n0 + nj) * CDIM + c0 + ci] = (bh_t)tile[ci][nj];
    }
}

__global__ __launch_bounds__(256) void k_cvt(const float* __restrict__ s, bh_t* __restrict__ d, int n) {
    int i = blockIdx.x * 256 + threadIdx.x;
    if (i < n) d[i] = (bh_t)s[i];
}

// ---------------- K1: QKV GEMM (384x128 @ 128x32 tile) + fused q-softmax ----------------
__global__ __launch_bounds__(256) void k1_qkv(const bh_t* __restrict__ xT, const bh_t* __restrict__ wq,
                                              bh_t* __restrict__ qws, float* __restrict__ kws,
                                              bh_t* __restrict__ vws) {
    __shared__ float qkvLDS[O3 * 33];
    __shared__ bh_t  xtile[32][TPITCH];
    const int b = blockIdx.y;
    const int n0 = blockIdx.x * 32;
    const int tid = threadIdx.x;
    const int wv = tid >> 5, lane = tid & 31;
    const int lm = lane & 15, hi = lane >> 4;

    // TDM: DMA the 32x128 bf16 x-tile into padded LDS
    stage_tile(xtile, xT + ((size_t)(b * NSEQ + n0)) * CDIM, 32, tid);
    __builtin_prefetch(xT + ((size_t)(b * NSEQ + n0 + 32)) * CDIM, 0, 1);
    __syncthreads();

    f32x8 acc[3][2];
#pragma unroll
    for (int i = 0; i < 3; ++i)
#pragma unroll
        for (int nt = 0; nt < 2; ++nt) acc[i][nt] = (f32x8)0.0f;

#pragma unroll
    for (int nt = 0; nt < 2; ++nt) {
#pragma unroll
        for (int kc = 0; kc < CDIM; kc += 32) {
            const bh_t* bp = &xtile[nt * 16 + lm][kc + hi * 8];
            bf16x16 bfr = ld16(bp, bp + 16);
#pragma unroll
            for (int i = 0; i < 3; ++i) {
                int row = (wv * 3 + i) * 16 + lm;
                const bh_t* ap = wq + (size_t)row * CDIM + kc + hi * 8;
                bf16x16 afr = ld16(ap, ap + 16);
                acc[i][nt] = wmma_bf16(afr, bfr, acc[i][nt]);
            }
        }
    }
    // stash D tiles: element (m,n) lives in lane (n%16)+(m>=8)*16, vgpr m%8
#pragma unroll
    for (int i = 0; i < 3; ++i)
#pragma unroll
        for (int nt = 0; nt < 2; ++nt)
#pragma unroll
            for (int r = 0; r < 8; ++r)
                qkvLDS[((wv * 3 + i) * 16 + r + hi * 8) * 33 + nt * 16 + lm] = acc[i][nt][r];
    __syncthreads();

    // q: softmax over dh (rows 0..127, per head) * dh^-0.5, write n-major bf16
    if (tid < 128) {
        int n = tid & 31, h = tid >> 5;
        float vals[32];
        float mx = -1e30f;
#pragma unroll
        for (int d = 0; d < 32; ++d) { vals[d] = qkvLDS[(h * 32 + d) * 33 + n]; mx = fmaxf(mx, vals[d]); }
        float s = 0.0f;
#pragma unroll
        for (int d = 0; d < 32; ++d) { vals[d] = __expf(vals[d] - mx); s += vals[d]; }
        float sc = 0.17677669529663687f / s;   // (1/sqrt(32)) / sum
        bh_t* qp = qws + ((size_t)(b * NSEQ + n0 + n)) * CDIM + h * 32;
#pragma unroll
        for (int d = 0; d < 32; ++d) qp[d] = (bh_t)(vals[d] * sc);
    }
    // k raw f32, v bf16 (rows 128..255 / 256..383), [b][hd][n] layout
    for (int idx = tid; idx < 128 * 32; idx += 256) {
        int r = idx >> 5, n = idx & 31;
        size_t o = ((size_t)(b * 128 + r)) * NSEQ + n0 + n;
        kws[o] = qkvLDS[(128 + r) * 33 + n];
        vws[o] = (bh_t)qkvLDS[(256 + r) * 33 + n];
    }
}

// ---------------- K2: per-(b,h,d) max and sum(exp) over n ----------------
__global__ __launch_bounds__(256) void k2_stats(const float* __restrict__ kws,
                                                float* __restrict__ kmax, float* __restrict__ ksum) {
    __shared__ float red[256];
    int row = blockIdx.x, tid = threadIdx.x;
    const float* p = kws + (size_t)row * NSEQ;
    float m = -1e30f;
    for (int j = tid; j < NSEQ; j += 256) m = fmaxf(m, p[j]);
    red[tid] = m; __syncthreads();
    for (int s = 128; s > 0; s >>= 1) { if (tid < s) red[tid] = fmaxf(red[tid], red[tid + s]); __syncthreads(); }
    float mx = red[0]; __syncthreads();
    float sum = 0.0f;
    for (int j = tid; j < NSEQ; j += 256) sum += __expf(p[j] - mx);
    red[tid] = sum; __syncthreads();
    for (int s = 128; s > 0; s >>= 1) { if (tid < s) red[tid] += red[tid + s]; __syncthreads(); }
    if (tid == 0) { kmax[row] = mx; ksum[row] = red[0]; }
}

// ---------------- K3: context = softmax_k @ v^T (32x32 per head), WMMA over K=16384 ----------------
__global__ __launch_bounds__(256) void k3_ctx(const float* __restrict__ kws, const bh_t* __restrict__ vws,
                                              const float* __restrict__ kmax, const float* __restrict__ ksum,
                                              bh_t* __restrict__ ctxT) {
    __shared__ float part[8][4][8][32];
    int bh = blockIdx.x;                       // b*4 + h
    int tid = threadIdx.x, wv = tid >> 5, lane = tid & 31, lm = lane & 15, hi = lane >> 4;
    int rowbase = bh * 32;                     // = b*128 + h*32

    f32x8 acc[2][2];
#pragma unroll
    for (int i = 0; i < 2; ++i)
#pragma unroll
        for (int j = 0; j < 2; ++j) acc[i][j] = (f32x8)0.0f;

    float mx[2], isum[2];
#pragma unroll
    for (int dt = 0; dt < 2; ++dt) {
        int d = dt * 16 + lm;
        mx[dt] = kmax[rowbase + d];
        isum[dt] = 1.0f / ksum[rowbase + d];
    }
    int nbase = wv * 2048;
    for (int s = 0; s < 64; ++s) {
        int n = nbase + s * 32;
        bf16x16 afr[2];
#pragma unroll
        for (int dt = 0; dt < 2; ++dt) {
            const float* p = kws + (size_t)(rowbase + dt * 16 + lm) * NSEQ + n + hi * 8;
            bf16x16 a;
#pragma unroll
            for (int j = 0; j < 8; ++j) a[j] = (bh_t)(__expf(p[j] - mx[dt]) * isum[dt]);
#pragma unroll
            for (int j = 0; j < 8; ++j) a[j + 8] = (bh_t)(__expf(p[j + 16] - mx[dt]) * isum[dt]);
            afr[dt] = a;
        }
        bf16x16 bfr[2];
#pragma unroll
        for (int et = 0; et < 2; ++et) {
            const bh_t* p = vws + (size_t)(rowbase + et * 16 + lm) * NSEQ + n + hi * 8;
            bfr[et] = ld16(p, p + 16);
        }
#pragma unroll
        for (int dt = 0; dt < 2; ++dt)
#pragma unroll
            for (int et = 0; et < 2; ++et) acc[dt][et] = wmma_bf16(afr[dt], bfr[et], acc[dt][et]);
    }
#pragma unroll
    for (int dt = 0; dt < 2; ++dt)
#pragma unroll
        for (int et = 0; et < 2; ++et)
#pragma unroll
            for (int r = 0; r < 8; ++r) part[wv][dt * 2 + et][r][lane] = acc[dt][et][r];
    __syncthreads();
    for (int idx = tid; idx < 1024; idx += 256) {
        int d = idx >> 5, e = idx & 31;
        int tile = (d >> 4) * 2 + (e >> 4);
        int m = d & 15, nn = e & 15;
        int ln = nn + ((m >= 8) ? 16 : 0);
        int r = m & 7;
        float sum = 0.0f;
#pragma unroll
        for (int w2 = 0; w2 < 8; ++w2) sum += part[w2][tile][r][ln];
        ctxT[((size_t)bh * 32 + e) * 32 + d] = (bh_t)sum;   // transposed: [bh][e][d]
    }
}

// ---------------- K4: out = ctx^T @ q  ->  y = W_out @ out + b, fused L2 norm ----------------
__global__ __launch_bounds__(256) void k4_out(const bh_t* __restrict__ ctxT, const bh_t* __restrict__ qws,
                                              const bh_t* __restrict__ wo, const float* __restrict__ bout,
                                              const float* __restrict__ g, float* __restrict__ out) {
    __shared__ bh_t  qtile[32][TPITCH];
    __shared__ bh_t  midT[32][136];
    __shared__ float yLDS[128][33];
    __shared__ float invn[32];
    const int b = blockIdx.y, n0 = blockIdx.x * 32;
    const int tid = threadIdx.x, wv = tid >> 5, lane = tid & 31, lm = lane & 15, hi = lane >> 4;

    // TDM: DMA the 32x128 bf16 q-tile (n-major) into padded LDS
    stage_tile(qtile, qws + ((size_t)(b * NSEQ + n0)) * CDIM, 32, tid);
    __syncthreads();

    // stage 1: per head, mid[e,n] = sum_d ctxT[e,d] * q[d,n]; K=32 -> 1 WMMA per tile
#pragma unroll
    for (int j = 0; j < 2; ++j) {
        int id = wv * 2 + j;
        int h = id >> 2, rem = id & 3, et = rem >> 1, nt = rem & 1;
        const bh_t* ap = ctxT + ((size_t)(b * 4 + h) * 32 + et * 16 + lm) * 32 + hi * 8;
        bf16x16 afr = ld16(ap, ap + 16);
        const bh_t* bp = &qtile[nt * 16 + lm][h * 32 + hi * 8];
        bf16x16 bfr = ld16(bp, bp + 16);
        f32x8 a = (f32x8)0.0f;
        a = wmma_bf16(afr, bfr, a);
#pragma unroll
        for (int r = 0; r < 8; ++r)
            midT[nt * 16 + lm][h * 32 + et * 16 + r + hi * 8] = (bh_t)a[r];
    }
    __syncthreads();

    // stage 2: y[o,n] = sum_e W_out[o,e] * mid[e,n]
    f32x8 acc2[2];
#pragma unroll
    for (int nt = 0; nt < 2; ++nt) acc2[nt] = (f32x8)0.0f;
#pragma unroll
    for (int kc = 0; kc < CDIM; kc += 32) {
        const bh_t* ap = wo + (size_t)(wv * 16 + lm) * CDIM + kc + hi * 8;
        bf16x16 afr = ld16(ap, ap + 16);
#pragma unroll
        for (int nt = 0; nt < 2; ++nt) {
            const bh_t* bp = &midT[nt * 16 + lm][kc + hi * 8];
            bf16x16 bfr = ld16(bp, bp + 16);
            acc2[nt] = wmma_bf16(afr, bfr, acc2[nt]);
        }
    }
#pragma unroll
    for (int nt = 0; nt < 2; ++nt)
#pragma unroll
        for (int r = 0; r < 8; ++r) yLDS[wv * 16 + r + hi * 8][nt * 16 + lm] = acc2[nt][r];
    __syncthreads();

    if (tid < 32) {
        int n = tid;
        float ss = 0.0f;
#pragma unroll 4
        for (int o = 0; o < 128; ++o) { float v = yLDS[o][n] + bout[o]; ss += v * v; }
        invn[n] = 11.313708498984761f / fmaxf(sqrtf(ss), 1e-12f);   // sqrt(128)/max(norm,eps)
    }
    __syncthreads();
    for (int idx = tid; idx < 4096; idx += 256) {
        int o = idx >> 5, n = idx & 31;
        out[((size_t)b * CDIM + o) * NSEQ + n0 + n] = (yLDS[o][n] + bout[o]) * invn[n] * g[o];
    }
}

// ---------------- launch ----------------
extern "C" void kernel_launch(void* const* d_in, const int* in_sizes, int n_in,
                              void* d_out, int out_size, void* d_ws, size_t ws_size,
                              hipStream_t stream) {
    (void)in_sizes; (void)n_in; (void)out_size; (void)ws_size;
    const float* x     = (const float*)d_in[0];
    const float* w_qkv = (const float*)d_in[1];
    const float* w_out = (const float*)d_in[2];
    const float* b_out = (const float*)d_in[3];
    const float* g     = (const float*)d_in[4];
    float* out = (float*)d_out;

    char* ws = (char*)d_ws;
    size_t off = 0;
    auto alloc = [&](size_t bytes) { void* p = ws + off; off += (bytes + 255) & ~(size_t)255; return p; };

    bh_t*  xT   = (bh_t*) alloc((size_t)NBATCH * NSEQ * CDIM * 2);
    bh_t*  wqb  = (bh_t*) alloc((size_t)O3 * CDIM * 2);
    bh_t*  wob  = (bh_t*) alloc((size_t)CDIM * CDIM * 2);
    bh_t*  qws  = (bh_t*) alloc((size_t)NBATCH * NSEQ * CDIM * 2);
    float* kws  = (float*)alloc((size_t)NBATCH * CDIM * NSEQ * 4);
    bh_t*  vws  = (bh_t*) alloc((size_t)NBATCH * CDIM * NSEQ * 2);
    float* kmax = (float*)alloc(1024 * 4);
    float* ksum = (float*)alloc(1024 * 4);
    bh_t*  ctxT = (bh_t*) alloc((size_t)NBATCH * 4 * 32 * 32 * 2);

    k_cvt_xT<<<dim3(NSEQ / 32, CDIM / 32, NBATCH), 256, 0, stream>>>(x, xT);
    k_cvt<<<dim3(192), 256, 0, stream>>>(w_qkv, wqb, O3 * CDIM);
    k_cvt<<<dim3(64),  256, 0, stream>>>(w_out, wob, CDIM * CDIM);
    k1_qkv<<<dim3(NSEQ / 32, NBATCH), 256, 0, stream>>>(xT, wqb, qws, kws, vws);
    k2_stats<<<dim3(NBATCH * CDIM), 256, 0, stream>>>(kws, kmax, ksum);
    k3_ctx<<<dim3(NBATCH * 4), 256, 0, stream>>>(kws, vws, kmax, ksum, ctxT);
    k4_out<<<dim3(NSEQ / 32, NBATCH), 256, 0, stream>>>(ctxT, qws, wob, b_out, g, out);
}
